// RightNormGraphConv_23459111371208
// MI455X (gfx1250) — matrix operand order repaired and checked by reading, not compiled, not run on hardware
//
#include <hip/hip_runtime.h>

// ---------------------------------------------------------------------------
// RightNormGraphConv for MI455X (gfx1250, wave32, WMMA)
//   h   = (feat @ W) * 1/max(outdeg(src),1)        [50000 x 128]
//   out = bias + scatter_sum(h[src] -> dst)
// GEMM uses split-bf16 (hi+lo) with v_wmma_f32_16x16x32_bf16:
//   a*b ~= ah*bh + ah*bl + al*bh   (error ~2^-16, fp32-class here)
// A strip is pre-split once per block into LDS (bank-conflict-free layout);
// steady-state inner loop is ds_load_b128 + global_load_b128 + 3x WMMA.
// ---------------------------------------------------------------------------

typedef __attribute__((ext_vector_type(16))) __bf16         v16bf;
typedef __attribute__((ext_vector_type(8)))  float          v8f;
typedef __attribute__((ext_vector_type(16))) unsigned short v16u;
typedef __attribute__((ext_vector_type(8)))  unsigned short v8u;

#define IN_F  256
#define OUT_F 128
#define A_PITCH 264   // u16 per LDS row: 256 + 8 pad -> 132 DWORDs, 132%64=4

// Truncation split of fp32 into two bf16 bit patterns (no cvt instructions).
__device__ __forceinline__ void split2bf(float a, unsigned short& hi, unsigned short& lo) {
    unsigned int ab = __float_as_uint(a);
    unsigned int hb = ab & 0xFFFF0000u;
    hi = (unsigned short)(ab >> 16);
    float rem = a - __uint_as_float(hb);
    lo = (unsigned short)(__float_as_uint(rem) >> 16);
}

// out[i,j] = bias[j]; deg = 0
__global__ __launch_bounds__(256) void rn_init_kernel(const float* __restrict__ bias,
                                                      float* __restrict__ out,
                                                      float* __restrict__ deg,
                                                      int totOut, int nNodes) {
    int i = blockIdx.x * 256 + threadIdx.x;
    if (i < totOut)  out[i] = bias[i & (OUT_F - 1)];
    if (i < nNodes)  deg[i] = 0.0f;
}

// Split weight [256,128] into bf16 hi/lo, stored transposed [col][k] so that
// B fragments are contiguous 16-byte loads.
__global__ __launch_bounds__(256) void rn_prepw_kernel(const float* __restrict__ w,
                                                       unsigned short* __restrict__ whT,
                                                       unsigned short* __restrict__ wlT) {
    int i = blockIdx.x * 256 + threadIdx.x;          // 0 .. 256*128-1
    if (i >= IN_F * OUT_F) return;
    int k = i >> 7;            // 0..255
    int c = i & (OUT_F - 1);   // 0..127
    unsigned short hi, lo;
    split2bf(w[i], hi, lo);
    whT[(size_t)c * IN_F + k] = hi;
    wlT[(size_t)c * IN_F + k] = lo;
}

__global__ __launch_bounds__(256) void rn_degree_kernel(const int* __restrict__ src,
                                                        float* __restrict__ deg, int nEdges) {
    int i = blockIdx.x * 256 + threadIdx.x;
    if (i < nEdges) atomicAdd(&deg[src[i]], 1.0f);
}

// deg[i] <- 1/max(deg[i],1), one v_rcp_f32 per node.
__global__ __launch_bounds__(256) void rn_norm_kernel(float* __restrict__ deg, int nNodes) {
    int i = blockIdx.x * 256 + threadIdx.x;
    if (i < nNodes) deg[i] = __builtin_amdgcn_rcpf(fmaxf(deg[i], 1.0f));
}

// h = (feat @ W) * norm.  One block = one 16-row strip; wave w -> N-tile w.
__global__ __launch_bounds__(256)
void rn_gemm_kernel(const float* __restrict__ feat,
                    const unsigned short* __restrict__ whT,
                    const unsigned short* __restrict__ wlT,
                    const float* __restrict__ norm,
                    float* __restrict__ h, int nNodes) {
    __shared__ unsigned short sAh[16][A_PITCH];
    __shared__ unsigned short sAl[16][A_PITCH];

    const int tid  = threadIdx.x;
    const int wave = tid >> 5;        // 0..7 == N tile
    const int lane = tid & 31;
    const int half = lane >> 4;       // 0/1
    const int l15  = lane & 15;
    const int mbase = blockIdx.x * 16;

    // ---- Cooperative A-strip load + one-time hi/lo split into LDS ----------
    {
        const int lrow = tid >> 4;                    // 0..15
        const int cb   = (tid & 15) * 16;             // 0,16,...,240
        int grow = mbase + lrow;
        if (grow >= nNodes) grow = nNodes - 1;        // clamp (M is tile-exact anyway)
        const float* ap = feat + (size_t)grow * IN_F + cb;
        float4 x0 = *(const float4*)(ap + 0);
        float4 x1 = *(const float4*)(ap + 4);
        float4 x2 = *(const float4*)(ap + 8);
        float4 x3 = *(const float4*)(ap + 12);
        float af[16] = {x0.x, x0.y, x0.z, x0.w, x1.x, x1.y, x1.z, x1.w,
                        x2.x, x2.y, x2.z, x2.w, x3.x, x3.y, x3.z, x3.w};
        v8u h0, h1, l0, l1;
#pragma unroll
        for (int i = 0; i < 8; ++i) {
            unsigned short hi, lo;
            split2bf(af[i], hi, lo);
            h0[i] = hi; l0[i] = lo;
            split2bf(af[i + 8], hi, lo);
            h1[i] = hi; l1[i] = lo;
        }
        *(v8u*)&sAh[lrow][cb]     = h0;
        *(v8u*)&sAh[lrow][cb + 8] = h1;
        *(v8u*)&sAl[lrow][cb]     = l0;
        *(v8u*)&sAl[lrow][cb + 8] = l1;
    }
    __syncthreads();

    // ---- Main K loop: A from LDS, B from global (L2-hot), 3 WMMAs/chunk ----
    const int bcol = wave * 16 + l15;
    const unsigned short* bhp = whT + (size_t)bcol * IN_F;
    const unsigned short* blp = wlT + (size_t)bcol * IN_F;

    v8f acc = {};

#pragma unroll 4
    for (int kc = 0; kc < 8; ++kc) {                 // K = 256 in chunks of 32
        const int kbA = kc * 32 + half * 16;         // A: elems 0..15 = kbA..+15 (contiguous)
        const int kbB = kc * 32 + half * 16;         // B: elems 0..15 = kbB..+15 (contiguous)

        v8u ah0 = *(const v8u*)&sAh[l15][kbA];
        v8u ah1 = *(const v8u*)&sAh[l15][kbA + 8];
        v8u al0 = *(const v8u*)&sAl[l15][kbA];
        v8u al1 = *(const v8u*)&sAl[l15][kbA + 8];

        v8u bh0 = *(const v8u*)(bhp + kbB);
        v8u bh1 = *(const v8u*)(bhp + kbB + 8);
        v8u bl0 = *(const v8u*)(blp + kbB);
        v8u bl1 = *(const v8u*)(blp + kbB + 8);

        v16u ahu = __builtin_shufflevector(ah0, ah1, 0, 1, 2, 3, 4, 5, 6, 7,
                                           8, 9, 10, 11, 12, 13, 14, 15);
        v16u alu = __builtin_shufflevector(al0, al1, 0, 1, 2, 3, 4, 5, 6, 7,
                                           8, 9, 10, 11, 12, 13, 14, 15);
        v16u bhu = __builtin_shufflevector(bh0, bh1, 0, 1, 2, 3, 4, 5, 6, 7,
                                           8, 9, 10, 11, 12, 13, 14, 15);
        v16u blu = __builtin_shufflevector(bl0, bl1, 0, 1, 2, 3, 4, 5, 6, 7,
                                           8, 9, 10, 11, 12, 13, 14, 15);
        v16bf ah  = __builtin_bit_cast(v16bf, ahu);
        v16bf al  = __builtin_bit_cast(v16bf, alu);
        v16bf bhv = __builtin_bit_cast(v16bf, bhu);
        v16bf blv = __builtin_bit_cast(v16bf, blu);

        acc = __builtin_amdgcn_wmma_f32_16x16x32_bf16(false, ah, false, bhv,
                                                      (short)0, acc, false, false);
        acc = __builtin_amdgcn_wmma_f32_16x16x32_bf16(false, ah, false, blv,
                                                      (short)0, acc, false, false);
        acc = __builtin_amdgcn_wmma_f32_16x16x32_bf16(false, al, false, bhv,
                                                      (short)0, acc, false, false);
    }

    const int col = wave * 16 + l15;
    if (mbase + 16 <= nNodes) {
        // Uniform fast path: 8 unguarded stores, no per-element exec juggling.
#pragma unroll
        for (int v = 0; v < 8; ++v) {
            int r = mbase + v + half * 8;            // C/D layout: VGPR v -> row v + 8*half
            h[(size_t)r * OUT_F + col] = acc[v] * norm[r];
        }
    } else {
#pragma unroll
        for (int v = 0; v < 8; ++v) {
            int r = mbase + v + half * 8;
            if (r < nNodes) h[(size_t)r * OUT_F + col] = acc[v] * norm[r];
        }
    }
}

// One wave per edge: lane handles 4 columns. h gather + out atomics are L2-resident.
__global__ __launch_bounds__(256)
void rn_scatter_kernel(const int* __restrict__ src, const int* __restrict__ dst,
                       const float* __restrict__ h, float* __restrict__ out, int nEdges) {
    long long t = (long long)blockIdx.x * 256 + threadIdx.x;
    int e = (int)(t >> 5);
    if (e >= nEdges) return;
    int lane = (int)(t & 31);
    int s = src[e];
    int d = dst[e];
    const float4 v = *(const float4*)(h + (size_t)s * OUT_F + lane * 4);
    float* o = out + (size_t)d * OUT_F + lane * 4;
    atomicAdd(o + 0, v.x);
    atomicAdd(o + 1, v.y);
    atomicAdd(o + 2, v.z);
    atomicAdd(o + 3, v.w);
}

extern "C" void kernel_launch(void* const* d_in, const int* in_sizes, int n_in,
                              void* d_out, int out_size, void* d_ws, size_t ws_size,
                              hipStream_t stream) {
    const float* feat   = (const float*)d_in[0];
    const float* weight = (const float*)d_in[1];
    const float* bias   = (const float*)d_in[2];
    const int*   src    = (const int*)d_in[3];
    const int*   dst    = (const int*)d_in[4];
    float*       out    = (float*)d_out;

    const int nNodes = in_sizes[0] / IN_F;   // 50000
    const int nEdges = in_sizes[3];          // 800000

    // Workspace layout (total ~26.2 MB):
    //   [0, 256KB)       deg/norm  (nNodes floats)
    //   [256KB, 512KB)   whT + wlT (bf16 bit patterns, 64KB each)
    //   [512KB, ...)     h         (nNodes * 128 floats)
    float*          deg = (float*)d_ws;
    unsigned short* whT = (unsigned short*)((char*)d_ws + (256 << 10));
    unsigned short* wlT = whT + (size_t)IN_F * OUT_F;
    float*          hbuf = (float*)((char*)d_ws + (512 << 10));

    const int totOut = nNodes * OUT_F;

    rn_init_kernel<<<(totOut + 255) / 256, 256, 0, stream>>>(bias, out, deg, totOut, nNodes);
    rn_prepw_kernel<<<(IN_F * OUT_F + 255) / 256, 256, 0, stream>>>(weight, whT, wlT);
    rn_degree_kernel<<<(nEdges + 255) / 256, 256, 0, stream>>>(src, deg, nEdges);
    rn_norm_kernel<<<(nNodes + 255) / 256, 256, 0, stream>>>(deg, nNodes);
    rn_gemm_kernel<<<(nNodes + 15) / 16, 256, 0, stream>>>(feat, whT, wlT, deg, hbuf, nNodes);

    long long scatterThreads = (long long)nEdges * 32;
    int scatterBlocks = (int)((scatterThreads + 255) / 256);
    rn_scatter_kernel<<<scatterBlocks, 256, 0, stream>>>(src, dst, hbuf, out, nEdges);
}
